// Attention_16415365005327
// MI455X (gfx1250) — matrix operand order repaired and checked by reading, not compiled, not run on hardware
//
#include <hip/hip_runtime.h>
#include <hip/hip_bf16.h>

typedef __attribute__((ext_vector_type(16))) __bf16 v16bf;
typedef __attribute__((ext_vector_type(8)))  __bf16 v8bf;
typedef __attribute__((ext_vector_type(8)))  float  v8f;

#define DIM   1024
#define HEADS 16
#define DHEAD 64
#define SEQ   2048
#define BATCH 4
#define ROWS  (BATCH*SEQ)   // 8192

// ---------------------------------------------------------------- helpers

__device__ __forceinline__ v8f wmma_bf16(v16bf a, v16bf b, v8f c) {
    // D = A(16x32 bf16) x B(32x16 bf16) + C(16x16 f32)
    return __builtin_amdgcn_wmma_f32_16x16x32_bf16(
        false, a, false, b, (short)0, c, false, false);
}

__device__ __forceinline__ v16bf join16(v8bf lo, v8bf hi) {
    v16bf r;
#pragma unroll
    for (int i = 0; i < 8; ++i) { r[i] = lo[i]; r[i + 8] = hi[i]; }
    return r;
}

// 16B global -> LDS async copy (ASYNCcnt path). VDST = LDS byte address,
// VADDR = 64-bit global address, SADDR = off (GV mode).
__device__ __forceinline__ void async_cp16(const __bf16* g, const __bf16* l) {
    unsigned lofs = (unsigned)(unsigned long long)(const void*)l; // low 32b = LDS offset
    unsigned long long ga = (unsigned long long)(const void*)g;
    asm volatile("global_load_async_to_lds_b128 %0, %1, off"
                 :: "v"(lofs), "v"(ga) : "memory");
}
__device__ __forceinline__ void wait_async_all() {
    asm volatile("s_wait_asynccnt 0" ::: "memory");
}

// ---------------------------------------------------------------- TDM (Tensor Data Mover)
// One wave issues a 2D tile DMA global->LDS, tracked with TENSORcnt.

#if __has_builtin(__builtin_amdgcn_tensor_load_to_lds)
#define HAVE_TDM 1
typedef __attribute__((ext_vector_type(4))) unsigned v4u;
typedef __attribute__((ext_vector_type(4))) int      v4i;
typedef __attribute__((ext_vector_type(8))) int      v8i;

// D# per CDNA5 ISA 8.3/8.4: group0 = {flags, lds_addr, global_addr lo,
// global_addr hi | type=2}; group1 = {data_size=2B, tensor_dim0/1, tile_dim0/1,
// dim0 stride}; groups 2/3 zero (2D tile). tile_w elems contiguous, tile_h rows.
// clang-23 toolchain: 6-arg builtin (g0, g1, g2, g3, g4pad, cpol).
__device__ __forceinline__ void tdm_load_2d_bf16(unsigned lds_off, const void* gaddr,
                                                 unsigned tile_w, unsigned tile_h,
                                                 unsigned stride_elems) {
    unsigned long long ga = (unsigned long long)gaddr;
    v4u g0 = {1u,                                   // count=1, user descriptor
              lds_off,                              // lds byte address
              (unsigned)ga,                         // global_addr[31:0]
              ((unsigned)(ga >> 32) & 0x01FFFFFFu) | (2u << 30)}; // [56:32] | type=2
    v8i g1 = {(int)(1u << 16),                      // wg_mask=0, data_size=1 (2B)
              (int)(tile_w << 16),                  // tensor_dim0[15:0]
              (int)((tile_w >> 16) | (tile_h << 16)), // tensor_dim0 hi | tensor_dim1 lo
              (int)((tile_h >> 16) | (tile_w << 16)), // tensor_dim1 hi | tile_dim0
              (int)tile_h,                          // tile_dim1 | tile_dim2=0
              (int)stride_elems,                    // tensor_dim0_stride[31:0]
              0, 0};
    v4i z4 = {0, 0, 0, 0};
    v8i z8 = {0, 0, 0, 0, 0, 0, 0, 0};
    __builtin_amdgcn_tensor_load_to_lds(g0, g1, z4, z4, z8, 0);
}
__device__ __forceinline__ void wait_tensor_all() {
#if __has_builtin(__builtin_amdgcn_s_wait_tensorcnt)
    __builtin_amdgcn_s_wait_tensorcnt((short)0);
#else
    asm volatile("s_wait_tensorcnt 0" ::: "memory");
#endif
}
#endif

// ---------------------------------------------------------------- LayerNorm -> bf16

__global__ void __launch_bounds__(256) ln_bf16(const float* __restrict__ x,
                                               const float* __restrict__ gamma,
                                               __bf16* __restrict__ xn) {
    int row = blockIdx.x;
    int t   = threadIdx.x;
    const float* xr = x + (size_t)row * DIM;
    float4 v = *(const float4*)(xr + t * 4);
    float s  = v.x + v.y + v.z + v.w;
    float s2 = v.x * v.x + v.y * v.y + v.z * v.z + v.w * v.w;
#pragma unroll
    for (int o = 16; o > 0; o >>= 1) {
        s  += __shfl_xor(s,  o, 32);
        s2 += __shfl_xor(s2, o, 32);
    }
    __shared__ float red[16];
    int wave = t >> 5, lane = t & 31;
    if (lane == 0) { red[wave] = s; red[8 + wave] = s2; }
    __syncthreads();
    float mu = 0.f, m2 = 0.f;
#pragma unroll
    for (int i = 0; i < 8; ++i) { mu += red[i]; m2 += red[8 + i]; }
    mu *= (1.0f / DIM);
    float var = m2 * (1.0f / DIM) - mu * mu;
    float inv = rsqrtf(var + 1e-5f);
    const float* g = gamma + t * 4;
    __bf16* o = xn + (size_t)row * DIM + t * 4;
    float vv[4] = {v.x, v.y, v.z, v.w};
#pragma unroll
    for (int j = 0; j < 4; ++j) o[j] = (__bf16)((vv[j] - mu) * inv * g[j]);
}

// ---------------------------------------------------------------- transpose + cvt
// W[K,N] f32 (row major) -> Wt[N,K] bf16 (row major). 32x32 tiles via LDS.

__global__ void __launch_bounds__(256) cvt_t_bf16(const float* __restrict__ W,
                                                  __bf16* __restrict__ Wt,
                                                  int K, int N) {
    __shared__ float tile[32][33];
    int bx = blockIdx.x * 32;            // n
    int by = blockIdx.y * 32;            // k
    int tx = threadIdx.x & 31, ty = threadIdx.x >> 5;
#pragma unroll
    for (int r = ty; r < 32; r += 8)
        tile[r][tx] = W[(size_t)(by + r) * N + bx + tx];
    __syncthreads();
#pragma unroll
    for (int r = ty; r < 32; r += 8)
        Wt[(size_t)(bx + r) * K + by + tx] = (__bf16)tile[tx][r];
}

// ---------------------------------------------------------------- WMMA GEMM
// C[M,N] = A[M,K] x B[K,N], with B supplied pre-transposed: Bt[N,K] (row major).
// LDS tile [n][kk] staged with contiguous 16B async copies, double buffered.
// OMODE: 0 = bf16 C[M,N], 1 = f32 C[M,N] + bias, 2 = bf16 transposed C^T[N,M].

template <int OMODE>
__global__ void __launch_bounds__(256)
gemm_bf16_wmma(const __bf16* __restrict__ A, const __bf16* __restrict__ Bt,
               __bf16* __restrict__ Cb, float* __restrict__ Cf,
               const float* __restrict__ bias, int M, int N, int K) {
    __shared__ __align__(64) __bf16 Bs[2][64 * 32]; // [n][kk]
    int t    = threadIdx.x;
    int wave = t >> 5, lane = t & 31;
    int hl   = lane >> 4, n = lane & 15;
    int row0 = blockIdx.y * 128 + wave * 16;
    int col0 = blockIdx.x * 64;
    int nl   = t >> 2, kg = (t & 3) * 8;
    const __bf16* bsrc = Bt + (size_t)(col0 + nl) * K + kg;

    async_cp16(bsrc, &Bs[0][nl * 32 + kg]);  // prologue: K-tile 0

    v8f acc[4] = {};
    const __bf16* Arow = A + (size_t)(row0 + n) * K;
    wait_async_all();
    __syncthreads();

    for (int k0 = 0; k0 < K; k0 += 32) {
        int p = (k0 >> 5) & 1;
        if (k0 + 32 < K)
            async_cp16(bsrc + k0 + 32, &Bs[1 - p][nl * 32 + kg]);
        v8bf alo = *(const v8bf*)(Arow + k0 + 16 * hl);
        v8bf ahi = *(const v8bf*)(Arow + k0 + 16 * hl + 8);
        v16bf a  = join16(alo, ahi);
        v16bf bfr[4];
#pragma unroll
        for (int dc = 0; dc < 4; ++dc)
            bfr[dc] = *(const v16bf*)&Bs[p][(dc * 16 + n) * 32 + 16 * hl];
#pragma unroll
        for (int dc = 0; dc < 4; ++dc)
            acc[dc] = wmma_bf16(a, bfr[dc], acc[dc]);
        wait_async_all();
        __syncthreads();
    }

#pragma unroll
    for (int dc = 0; dc < 4; ++dc) {
        int col = col0 + dc * 16 + n;
        if constexpr (OMODE == 2) {
            v8bf ov;
#pragma unroll
            for (int r = 0; r < 8; ++r) ov[r] = (__bf16)acc[dc][r];
            *(v8bf*)(Cb + (size_t)col * M + row0 + 8 * hl) = ov; // contiguous rows
        } else {
#pragma unroll
            for (int r = 0; r < 8; ++r) {
                int row = row0 + r + 8 * hl;
                if constexpr (OMODE == 0) Cb[(size_t)row * N + col] = (__bf16)acc[dc][r];
                else                      Cf[(size_t)row * N + col] = acc[dc][r] + bias[col];
            }
        }
    }
}

// ---------------------------------------------------------------- Flash attention
// S^T = K x Q^T, O^T = V^T x P^T; per-lane scalar softmax stats (one xor-16
// shuffle per reduction). K [j][d] and V^T [d][j] tiles are double-buffered in
// LDS, staged by the Tensor Data Mover (wave0 -> K desc, wave1 -> V^T desc,
// TENSORcnt) when available, else per-lane async copies (ASYNCcnt).

__global__ void __launch_bounds__(256)
flash_attn_bf16(const __bf16* __restrict__ q, const __bf16* __restrict__ k,
                const __bf16* __restrict__ vt, __bf16* __restrict__ out) {
    __shared__ __align__(64) __bf16 ks[2][32 * 64];  // K tile  [j][d]
    __shared__ __align__(64) __bf16 vs[2][64 * 32];  // V^T tile [d][j]
    __shared__ __align__(64) __bf16 PT[8][16 * 48];  // per-wave P^T [i][j]
    const float scale = 0.125f; // DHEAD^-0.5, folded into the softmax exponent

    int t    = threadIdx.x;
    int wave = t >> 5, lane = t & 31;
    int hl   = lane >> 4, n = lane & 15;
    int b    = blockIdx.x >> 4, h = blockIdx.x & 15;
    int row0 = blockIdx.y * 128 + wave * 16;

    // Q as B fragments: contiguous 32B per fragment, loaded once
    const __bf16* qrow = q + (size_t)(b * SEQ + row0 + n) * DIM + h * DHEAD;
    v16bf qB[2];
#pragma unroll
    for (int dc = 0; dc < 2; ++dc)
        qB[dc] = *(const v16bf*)(qrow + dc * 32 + 16 * hl);

    int jl = t >> 3, dgk = (t & 7) * 8;  // fallback K staging:  [jl][dgk..+8)
    int dl = t >> 2, jg  = (t & 3) * 8;  // fallback V^T staging: [dl][jg..+8)
    const __bf16* kbase  = k  + (size_t)b * SEQ * DIM + h * DHEAD;
    const __bf16* vtbase = vt + (size_t)(h * DHEAD) * ROWS + (size_t)b * SEQ;

    auto stage_tiles = [&](int j0, int buf) {
#ifdef HAVE_TDM
        if (wave == 0)
            tdm_load_2d_bf16((unsigned)(unsigned long long)(const void*)&ks[buf][0],
                             kbase + (size_t)j0 * DIM, 64u, 32u, (unsigned)DIM);
        else if (wave == 1)
            tdm_load_2d_bf16((unsigned)(unsigned long long)(const void*)&vs[buf][0],
                             vtbase + j0, 32u, 64u, (unsigned)ROWS);
#else
        async_cp16(kbase + (size_t)(j0 + jl) * DIM + dgk, &ks[buf][jl * 64 + dgk]);
        async_cp16(vtbase + (size_t)dl * ROWS + j0 + jg, &vs[buf][dl * 32 + jg]);
#endif
    };
    auto wait_stage = [&]() {
#ifdef HAVE_TDM
        if (wave < 2) wait_tensor_all();
#else
        wait_async_all();
#endif
    };

    stage_tiles(0, 0);

    v8f oT[4] = {};
    float m_ = -3.0e38f, l_ = 0.0f;
    __bf16* Pw = &PT[wave][0];
    wait_stage();
    __syncthreads();

    for (int j0 = 0; j0 < SEQ; j0 += 32) {
        int p = (j0 >> 5) & 1;
        if (j0 + 32 < SEQ) stage_tiles(j0 + 32, 1 - p);

        // preload all K fragments, then issue the WMMA group
        v16bf kA[2][2];
#pragma unroll
        for (int jt = 0; jt < 2; ++jt)
#pragma unroll
            for (int dc = 0; dc < 2; ++dc) {
                const __bf16* kr = &ks[p][(jt * 16 + n) * 64 + dc * 32];
                kA[jt][dc] = join16(*(const v8bf*)(kr + 16 * hl),
                                    *(const v8bf*)(kr + 16 * hl + 8));
            }
        v8f st[2];
#pragma unroll
        for (int jt = 0; jt < 2; ++jt) {
            v8f accs = {};
            accs = wmma_bf16(kA[jt][0], qB[0], accs);
            accs = wmma_bf16(kA[jt][1], qB[1], accs);
            st[jt] = accs;  // raw scores; scale folded into exp below
        }

        // per-lane online softmax for q-row i = n (halves hold disjoint j)
        float mx = fmaxf(st[0][0], st[1][0]);
#pragma unroll
        for (int r = 1; r < 8; ++r) mx = fmaxf(mx, fmaxf(st[0][r], st[1][r]));
        mx = fmaxf(mx, __shfl_xor(mx, 16, 32));
        float mn    = fmaxf(m_, mx);
        float alpha = __expf((m_ - mn) * scale);
        float nms   = -mn * scale;
        float rs = 0.f;
#pragma unroll
        for (int jt = 0; jt < 2; ++jt)
#pragma unroll
            for (int r = 0; r < 8; ++r) {
                float pv = __expf(__builtin_fmaf(st[jt][r], scale, nms));
                st[jt][r] = pv;
                rs += pv;
            }
        rs += __shfl_xor(rs, 16, 32);
        l_ = l_ * alpha + rs;
        m_ = mn;

        // P^T C-layout -> B-layout via per-wave LDS bounce ([i][j], stride 48)
#pragma unroll
        for (int jt = 0; jt < 2; ++jt)
#pragma unroll
            for (int r = 0; r < 8; ++r)
                Pw[n * 48 + jt * 16 + r + 8 * hl] = (__bf16)st[jt][r];
        asm volatile("s_wait_dscnt 0" ::: "memory");
        v16bf pB = *(const v16bf*)(Pw + n * 48 + 16 * hl);

        // preload V^T fragments, then O^T = O^T*alpha + V^T x P^T
        v16bf vA[4];
#pragma unroll
        for (int dc4 = 0; dc4 < 4; ++dc4) {
            const __bf16* vr = &vs[p][(dc4 * 16 + n) * 32];
            vA[dc4] = join16(*(const v8bf*)(vr + 16 * hl),
                             *(const v8bf*)(vr + 16 * hl + 8));
        }
#pragma unroll
        for (int dc4 = 0; dc4 < 4; ++dc4) {
            v8f oo = oT[dc4] * alpha;
            oT[dc4] = wmma_bf16(vA[dc4], pB, oo);
        }
        wait_stage();
        __syncthreads();
    }

    // normalize; O^T cols are rows i=n, rows are d -> contiguous 16B stores
    float linv = 1.0f / l_;
#pragma unroll
    for (int dc4 = 0; dc4 < 4; ++dc4) {
        v8bf ov;
#pragma unroll
        for (int r = 0; r < 8; ++r) ov[r] = (__bf16)(oT[dc4][r] * linv);
        *(v8bf*)(out + (size_t)(b * SEQ + row0 + n) * DIM + h * DHEAD + dc4 * 16 + 8 * hl) = ov;
    }
}

// ---------------------------------------------------------------- launch

extern "C" void kernel_launch(void* const* d_in, const int* in_sizes, int n_in,
                              void* d_out, int out_size, void* d_ws, size_t ws_size,
                              hipStream_t stream) {
    const float* x     = (const float*)d_in[0];
    const float* gamma = (const float*)d_in[1];
    const float* Wq    = (const float*)d_in[2];
    const float* Wkv   = (const float*)d_in[3];
    const float* Wout  = (const float*)d_in[4];
    const float* bout  = (const float*)d_in[5];
    float* outp = (float*)d_out;

    char* ws = (char*)d_ws;
    const size_t MB = 1024u * 1024u;
    __bf16* xnb  = (__bf16*)(ws + 0);        // 16 MB (LN out; reused for attn out)
    __bf16* Wqt  = (__bf16*)(ws + 16 * MB);  // 2 MB  [N=1024][K=1024]
    __bf16* Wkvt = (__bf16*)(ws + 18 * MB);  // 4 MB  [N=2048][K=1024]
    __bf16* Wot  = (__bf16*)(ws + 22 * MB);  // 2 MB
    __bf16* qb   = (__bf16*)(ws + 24 * MB);  // 16 MB [8192][1024]
    __bf16* kb   = (__bf16*)(ws + 40 * MB);  // 16 MB [8192][1024]
    __bf16* vtb  = (__bf16*)(ws + 56 * MB);  // 16 MB transposed: [1024][8192]
    __bf16* aob  = xnb;                      // xn dead after projections

    dim3 cb(256);
    cvt_t_bf16<<<dim3(DIM / 32, DIM / 32), cb, 0, stream>>>(Wq, Wqt, DIM, DIM);
    cvt_t_bf16<<<dim3(2048 / 32, DIM / 32), cb, 0, stream>>>(Wkv, Wkvt, DIM, 2048);
    cvt_t_bf16<<<dim3(DIM / 32, DIM / 32), cb, 0, stream>>>(Wout, Wot, DIM, DIM);

    ln_bf16<<<ROWS, 256, 0, stream>>>(x, gamma, xnb);

    dim3 g1(DIM / 64, ROWS / 128);
    // Q projection (normal bf16 out)
    gemm_bf16_wmma<0><<<g1, 256, 0, stream>>>(xnb, Wqt, qb, nullptr, nullptr, ROWS, DIM, DIM);
    // K projection: Wkv columns 0..1023 == Wkvt rows 0..1023
    gemm_bf16_wmma<0><<<g1, 256, 0, stream>>>(xnb, Wkvt, kb, nullptr, nullptr, ROWS, DIM, DIM);
    // V projection: Wkv columns 1024..2047; store transposed [col][row]
    gemm_bf16_wmma<2><<<g1, 256, 0, stream>>>(xnb, Wkvt + (size_t)1024 * DIM, vtb, nullptr,
                                              nullptr, ROWS, DIM, DIM);

    dim3 ga(BATCH * HEADS, SEQ / 128);
    flash_attn_bf16<<<ga, 256, 0, stream>>>(qb, kb, vtb, aob);

    gemm_bf16_wmma<1><<<g1, 256, 0, stream>>>(aob, Wot, nullptr, outp, bout, ROWS, DIM, DIM);
}
